// AdvancedNeuralMemory_64518998721112
// MI455X (gfx1250) — compile-verified, matrix operand order checked by reading, not run on hardware
//
#include <hip/hip_runtime.h>
#include <hip/hip_bf16.h>
#include <math.h>
#include <stdint.h>

typedef __attribute__((ext_vector_type(16))) _Float16     v16h;
typedef __attribute__((ext_vector_type(8)))  float        v8f;
typedef __attribute__((ext_vector_type(4)))  unsigned int v4u;
typedef __attribute__((ext_vector_type(4)))  int          v4i;

#define EPI_F32    0
#define EPI_F16    1
#define EPI_GELU16 2

#if defined(__gfx1250__) && __has_builtin(__builtin_amdgcn_global_load_async_to_lds_b128) && __has_builtin(__builtin_amdgcn_s_wait_asynccnt)
#define HAVE_ASYNC_LDS 1
#else
#define HAVE_ASYNC_LDS 0
#endif

#if HAVE_ASYNC_LDS
typedef __attribute__((address_space(1))) v4i* g_v4i;
typedef __attribute__((address_space(3))) v4i* l_v4i;

__device__ __forceinline__ g_v4i to_global_v4i(const void* p) {
  // generic->global: identical bit pattern
  return (g_v4i)(unsigned long long)(uintptr_t)p;
}
__device__ __forceinline__ l_v4i to_lds_v4i(const void* p) {
  // generic LDS pointer: low 32 bits are the LDS offset (ISA aperture rule)
  return (l_v4i)(unsigned int)(uintptr_t)p;
}
#endif

__device__ __forceinline__ float gelu_f(float x) {
  const float c = 0.7978845608028654f; // sqrt(2/pi)
  return 0.5f * x * (1.0f + tanhf(c * (x + 0.044715f * x * x * x)));
}

__device__ __forceinline__ float gate_of(const float* ff) {
  return 1.0f - 1.0f / (1.0f + expf(-ff[0]));
}

// C[M][N] = A[M][K] (f16, or f32 converted while staging) * BT[N][K]^T (f16),
// f32 accumulate via v_wmma_f32_16x16x32_f16.
// 128 threads = 4 waves. Block tile 128x128, K-step 32, double-buffered LDS.
// Wave tile 64x64 => 4x4 WMMA per K-step (16 wmma per 16 frag ds_load_b128).
// f16 tiles are staged with GLOBAL_LOAD_ASYNC_TO_LDS_B128 when available.
template<int EPI, int A_IS_F32>
__global__ __launch_bounds__(128) void gemm_wmma_kernel(
    const void* __restrict__ Av, const _Float16* __restrict__ BT,
    const float* __restrict__ bias, void* __restrict__ Cv,
    int M, int N, int K)
{
  constexpr int BM = 128, BN = 128, BK = 32, LDT = BK + 8; // pad keeps 16B align
  __shared__ _Float16 As[2][BM * LDT];
  __shared__ _Float16 Bs[2][BN * LDT];

  const int tid   = threadIdx.x;
  const int wave  = tid >> 5;
  const int lane  = tid & 31;
  const int mBase = blockIdx.x * BM;
  const int nBase = blockIdx.y * BN;
  const int wm    = (wave >> 1) * 64;
  const int wn    = (wave & 1) * 64;
  const int hih   = lane >> 4;   // K sub-block select within fragment
  const int l16   = lane & 15;   // row/col within 16

  const float*    Af = (const float*)Av;
  const _Float16* Ah = (const _Float16*)Av;

  v8f acc[4][4] = {};

  auto stageA = [&](int buf, int k0) {
    if (A_IS_F32) {
      // one 128-row tile; one row (32 floats -> 32 halves) per thread
      const float* src = Af + (size_t)(mBase + tid) * K + k0;
      if (k0 + BK < K)
        __builtin_prefetch((const void*)(src + BK), 0, 1);
      _Float16* dst = &As[buf][tid * LDT];
      #pragma unroll
      for (int u = 0; u < 32; ++u) dst[u] = (_Float16)src[u];
    } else {
#if HAVE_ASYNC_LDS
      #pragma unroll
      for (int c = 0; c < 4; ++c) {           // 512 x 16B units, 128 lanes
        const int u   = c * 128 + tid;
        const int row = u >> 2;
        const int seg = (u & 3) * 8;          // halves
        __builtin_amdgcn_global_load_async_to_lds_b128(
            to_global_v4i(Ah + (size_t)(mBase + row) * K + k0 + seg),
            to_lds_v4i(&As[buf][row * LDT + seg]),
            0, 0);
      }
#else
      const v4u* src = (const v4u*)(Ah + (size_t)(mBase + tid) * K + k0);
      v4u* dst = (v4u*)&As[buf][tid * LDT];
      #pragma unroll
      for (int c = 0; c < 4; ++c) dst[c] = src[c];
#endif
    }
  };
  auto stageB = [&](int buf, int k0) {
#if HAVE_ASYNC_LDS
    #pragma unroll
    for (int c = 0; c < 4; ++c) {
      const int u   = c * 128 + tid;
      const int row = u >> 2;
      const int seg = (u & 3) * 8;
      __builtin_amdgcn_global_load_async_to_lds_b128(
          to_global_v4i(BT + (size_t)(nBase + row) * K + k0 + seg),
          to_lds_v4i(&Bs[buf][row * LDT + seg]),
          0, 0);
    }
#else
    const v4u* src = (const v4u*)(BT + (size_t)(nBase + tid) * K + k0);
    v4u* dst = (v4u*)&Bs[buf][tid * LDT];
    #pragma unroll
    for (int c = 0; c < 4; ++c) dst[c] = src[c];
#endif
  };

  stageA(0, 0);
  stageB(0, 0);
#if HAVE_ASYNC_LDS
  __builtin_amdgcn_s_wait_asynccnt(0);
#endif
  __syncthreads();

  int cur = 0;
  for (int k0 = 0; k0 < K; k0 += BK) {
    if (k0 + BK < K) {           // stage next tile into the other buffer
      stageA(cur ^ 1, k0 + BK);
      stageB(cur ^ 1, k0 + BK);
    }

    // Fragment layout (16-bit A 16x32): lane l -> row l%16;
    // lanes 0-15 hold K = {0..7, 16..23}; lanes 16-31 hold K = {8..15, 24..31}.
    union Frag { v16h v; v4u q[2]; };
    Frag a[4], b[4];
    #pragma unroll
    for (int i = 0; i < 4; ++i) {
      const _Float16* pa = &As[cur][(wm + i * 16 + l16) * LDT + hih * 8];
      a[i].q[0] = *(const v4u*)pa;
      a[i].q[1] = *(const v4u*)(pa + 16);
      const _Float16* pb = &Bs[cur][(wn + i * 16 + l16) * LDT + hih * 8];
      b[i].q[0] = *(const v4u*)pb;
      b[i].q[1] = *(const v4u*)(pb + 16);
    }
    #pragma unroll
    for (int i = 0; i < 4; ++i)
      #pragma unroll
      for (int j = 0; j < 4; ++j)
        acc[i][j] = __builtin_amdgcn_wmma_f32_16x16x32_f16(
            false, a[i].v, false, b[j].v, (short)0, acc[i][j], false, false);

#if HAVE_ASYNC_LDS
    __builtin_amdgcn_s_wait_asynccnt(0);
#endif
    __syncthreads();
    cur ^= 1;
  }

  // Epilogue. C layout: VGPR e -> row = e + (lane/16)*8, col = lane%16.
  #pragma unroll
  for (int i = 0; i < 4; ++i) {
    #pragma unroll
    for (int j = 0; j < 4; ++j) {
      const int gn = nBase + wn + j * 16 + l16;
      const float bb = bias ? bias[gn] : 0.0f;
      #pragma unroll
      for (int e = 0; e < 8; ++e) {
        const int gm = mBase + wm + i * 16 + hih * 8 + e;
        float val = acc[i][j][e] + bb;
        if (EPI == EPI_GELU16) val = gelu_f(val);
        if (EPI == EPI_F32) ((float*)Cv)[(size_t)gm * N + gn] = val;
        else               ((_Float16*)Cv)[(size_t)gm * N + gn] = (_Float16)val;
      }
    }
  }
}

// Cast fp32 weight [K][N] to f16 transposed [N][K].
__global__ void cast_transpose_kernel(const float* __restrict__ W,
                                      _Float16* __restrict__ WT, int K, int N)
{
  size_t id = (size_t)blockIdx.x * blockDim.x + threadIdx.x;
  if (id >= (size_t)K * N) return;
  int k = (int)(id / N), n = (int)(id % N);
  WT[(size_t)n * K + k] = (_Float16)W[id];
}

// LayerNorm over 512 features, one wave (32 lanes) per row, f32 in -> f16 out.
__global__ __launch_bounds__(128) void ln512_kernel(
    const float* __restrict__ X, const float* __restrict__ gamma,
    const float* __restrict__ beta, _Float16* __restrict__ Y, int rows)
{
  const int lane = threadIdx.x & 31;
  const int row  = blockIdx.x * 4 + (threadIdx.x >> 5);
  if (row >= rows) return;
  const float* x = X + (size_t)row * 512;
  float v[16]; float s = 0.f;
  #pragma unroll
  for (int i = 0; i < 16; ++i) { v[i] = x[i * 32 + lane]; s += v[i]; }
  #pragma unroll
  for (int o = 16; o > 0; o >>= 1) s += __shfl_xor(s, o, 32);
  const float mean = s * (1.0f / 512.0f);
  float q = 0.f;
  #pragma unroll
  for (int i = 0; i < 16; ++i) { float d = v[i] - mean; q += d * d; }
  #pragma unroll
  for (int o = 16; o > 0; o >>= 1) q += __shfl_xor(q, o, 32);
  const float rstd = rsqrtf(q * (1.0f / 512.0f) + 1e-5f);
  #pragma unroll
  for (int i = 0; i < 16; ++i) {
    int c = i * 32 + lane;
    Y[(size_t)row * 512 + c] = (_Float16)((v[i] - mean) * rstd * gamma[c] + beta[c]);
  }
}

// surprise = mean((pred-v)^2); io_pred row overwritten with lr*surprise*v.
__global__ __launch_bounds__(128) void surprise512_kernel(
    float* __restrict__ io_pred, const float* __restrict__ V,
    const float* __restrict__ lr_p, int rows)
{
  const int lane = threadIdx.x & 31;
  const int row  = blockIdx.x * 4 + (threadIdx.x >> 5);
  if (row >= rows) return;
  float* p = io_pred + (size_t)row * 512;
  const float* v = V + (size_t)row * 512;
  float vv[16]; float s = 0.f;
  #pragma unroll
  for (int i = 0; i < 16; ++i) {
    vv[i] = v[i * 32 + lane];
    float d = p[i * 32 + lane] - vv[i];
    s += d * d;
  }
  #pragma unroll
  for (int o = 16; o > 0; o >>= 1) s += __shfl_xor(s, o, 32);
  const float wmul = lr_p[0] * (s * (1.0f / 512.0f));
  #pragma unroll
  for (int i = 0; i < 16; ++i) p[i * 32 + lane] = wmul * vv[i];
}

// --- chunked EMA scan (gate is constant per launch) ---
// Pass 1: local scans of CH steps per (b, chunk, d); carry = last local value.
__global__ void scan_local_kernel(float* __restrict__ io, float* __restrict__ carry,
                                  const float* __restrict__ ff,
                                  int Bn, int S, int C, int CH)
{
  const int NCH = S / CH;
  const int j = blockIdx.x * blockDim.x + threadIdx.x;
  if (j >= Bn * NCH * C) return;
  const int d = j % C;
  const int r = j / C;
  const int c = r % NCH;
  const int b = r / NCH;
  const float g = gate_of(ff);
  float* p = io + ((size_t)b * S + (size_t)c * CH) * C + d;
  float s = 0.f;
  for (int t = 0; t < CH; ++t) { s = g * s + p[(size_t)t * C]; p[(size_t)t * C] = s; }
  carry[((size_t)b * NCH + c) * C + d] = s;
}

// Pass 2: scan the chunk carries: E_c = carry_c + g^CH * E_{c-1}; store E_{c-1}.
__global__ void scan_carry_kernel(const float* __restrict__ carry,
                                  float* __restrict__ epref,
                                  const float* __restrict__ ff,
                                  int Bn, int NCH, int C, int CH)
{
  const int j = blockIdx.x * blockDim.x + threadIdx.x;
  if (j >= Bn * C) return;
  const int b = j / C, d = j % C;
  const float g  = gate_of(ff);
  const float gL = powf(g, (float)CH);
  float E = 0.f;
  for (int c = 0; c < NCH; ++c) {
    const size_t idx = ((size_t)b * NCH + c) * C + d;
    epref[idx] = E;
    E = carry[idx] + gL * E;
  }
}

// Fused fix-up + combine + cast: rs16 = f16(ret + local + E_{c-1} * g^(tloc+1)).
__global__ void combine_fix_kernel(const float* __restrict__ ret,
                                   const float* __restrict__ loc,
                                   const float* __restrict__ epref,
                                   const float* __restrict__ ff,
                                   _Float16* __restrict__ y,
                                   int Bn, int S, int C, int CH)
{
  const size_t n = (size_t)Bn * S * C;
  const int NCH = S / CH;
  const float g = gate_of(ff);
  size_t i = (size_t)blockIdx.x * blockDim.x + threadIdx.x;
  const size_t stride = (size_t)gridDim.x * blockDim.x;
  for (; i < n; i += stride) {
    const int d = (int)(i % C);
    const size_t r = i / C;
    const int t = (int)(r % S);
    const int b = (int)(r / S);
    const int c = t / CH, tl = t - c * CH;
    const float corr = epref[((size_t)b * NCH + c) * C + d] * powf(g, (float)(tl + 1));
    y[i] = (_Float16)(ret[i] + loc[i] + corr);
  }
}

extern "C" void kernel_launch(void* const* d_in, const int* in_sizes, int n_in,
                              void* d_out, int out_size, void* d_ws, size_t ws_size,
                              hipStream_t stream)
{
  (void)in_sizes; (void)n_in; (void)out_size; (void)ws_size;
  constexpr int Bn = 4, S = 4096, D = 1024, MD = 512;
  constexpr int NT = Bn * S;          // 16384 tokens
  constexpr int CH = 128, NCH = S / CH;

  const float* x  = (const float*)d_in[0];
  const float* Wd = (const float*)d_in[1];
  const float* bd = (const float*)d_in[2];
  const float* Wu = (const float*)d_in[3];
  const float* bu = (const float*)d_in[4];
  const float* Wq = (const float*)d_in[5];
  const float* bq = (const float*)d_in[6];
  const float* Wk = (const float*)d_in[7];
  const float* bk = (const float*)d_in[8];
  const float* Wv = (const float*)d_in[9];
  const float* bv = (const float*)d_in[10];
  const float* qg = (const float*)d_in[11];
  const float* qb = (const float*)d_in[12];
  const float* kg = (const float*)d_in[13];
  const float* kb = (const float*)d_in[14];
  const float* W0 = (const float*)d_in[15];
  const float* W1 = (const float*)d_in[16];
  const float* lr = (const float*)d_in[17];
  const float* ff = (const float*)d_in[18];

  char* wsp = (char*)d_ws;
  auto alloc = [&](size_t bytes) -> void* {
    void* r = (void*)wsp;
    wsp += (bytes + 255) & ~(size_t)255;
    return r;
  };
  _Float16* WdT   = (_Float16*)alloc((size_t)D * MD * 2);
  _Float16* WqT   = (_Float16*)alloc((size_t)MD * MD * 2);
  _Float16* WkT   = (_Float16*)alloc((size_t)MD * MD * 2);
  _Float16* WvT   = (_Float16*)alloc((size_t)MD * MD * 2);
  _Float16* W0T   = (_Float16*)alloc((size_t)MD * MD * 2);
  _Float16* W1T   = (_Float16*)alloc((size_t)MD * MD * 2);
  _Float16* WuT   = (_Float16*)alloc((size_t)MD * D * 2);
  _Float16* h16   = (_Float16*)alloc((size_t)NT * MD * 2);
  float*    pre32 = (float*)   alloc((size_t)NT * MD * 4);
  _Float16* q16   = (_Float16*)alloc((size_t)NT * MD * 2);
  _Float16* k16   = (_Float16*)alloc((size_t)NT * MD * 2);
  float*    v32   = (float*)   alloc((size_t)NT * MD * 4);
  _Float16* t16   = (_Float16*)alloc((size_t)NT * MD * 2);
  float*    ret32 = (float*)   alloc((size_t)NT * MD * 4);
  float*    pred32= (float*)   alloc((size_t)NT * MD * 4);
  _Float16* rs16  = (_Float16*)alloc((size_t)NT * MD * 2);
  float*    carry = (float*)   alloc((size_t)Bn * NCH * MD * 4);
  float*    epref = (float*)   alloc((size_t)Bn * NCH * MD * 4);

  auto ctg = [](size_t n) { return (unsigned)((n + 255) / 256); };
  cast_transpose_kernel<<<ctg((size_t)D * MD), 256, 0, stream>>>(Wd, WdT, D, MD);
  cast_transpose_kernel<<<ctg((size_t)MD * MD), 256, 0, stream>>>(Wq, WqT, MD, MD);
  cast_transpose_kernel<<<ctg((size_t)MD * MD), 256, 0, stream>>>(Wk, WkT, MD, MD);
  cast_transpose_kernel<<<ctg((size_t)MD * MD), 256, 0, stream>>>(Wv, WvT, MD, MD);
  cast_transpose_kernel<<<ctg((size_t)MD * MD), 256, 0, stream>>>(W0, W0T, MD, MD);
  cast_transpose_kernel<<<ctg((size_t)MD * MD), 256, 0, stream>>>(W1, W1T, MD, MD);
  cast_transpose_kernel<<<ctg((size_t)MD * D), 256, 0, stream>>>(Wu, WuT, MD, D);

  dim3 blk(128);
  dim3 g_md(NT / 128, MD / 128);  // 128 x 4
  dim3 g_d (NT / 128, D  / 128);  // 128 x 8

  // h = cast16(x @ Wd + bd)
  gemm_wmma_kernel<EPI_F16, 1><<<g_md, blk, 0, stream>>>(x, WdT, bd, h16, NT, MD, D);
  // q = LN(h @ Wq + bq)
  gemm_wmma_kernel<EPI_F32, 0><<<g_md, blk, 0, stream>>>(h16, WqT, bq, pre32, NT, MD, MD);
  ln512_kernel<<<NT / 4, 128, 0, stream>>>(pre32, qg, qb, q16, NT);
  // k = LN(h @ Wk + bk)
  gemm_wmma_kernel<EPI_F32, 0><<<g_md, blk, 0, stream>>>(h16, WkT, bk, pre32, NT, MD, MD);
  ln512_kernel<<<NT / 4, 128, 0, stream>>>(pre32, kg, kb, k16, NT);
  // v = h @ Wv + bv (kept f32)
  gemm_wmma_kernel<EPI_F32, 0><<<g_md, blk, 0, stream>>>(h16, WvT, bv, v32, NT, MD, MD);
  // retrieved = gelu(q @ W0) @ W1
  gemm_wmma_kernel<EPI_GELU16, 0><<<g_md, blk, 0, stream>>>(q16, W0T, nullptr, t16, NT, MD, MD);
  gemm_wmma_kernel<EPI_F32, 0><<<g_md, blk, 0, stream>>>(t16, W1T, nullptr, ret32, NT, MD, MD);
  // pred = gelu(k @ W0) @ W1
  gemm_wmma_kernel<EPI_GELU16, 0><<<g_md, blk, 0, stream>>>(k16, W0T, nullptr, t16, NT, MD, MD);
  gemm_wmma_kernel<EPI_F32, 0><<<g_md, blk, 0, stream>>>(t16, W1T, nullptr, pred32, NT, MD, MD);
  // inputs = lr * mean((pred-v)^2) * v   (overwrites pred32)
  surprise512_kernel<<<NT / 4, 128, 0, stream>>>(pred32, v32, lr, NT);
  // scanned = chunked EMA scan (local in place + carries + fused fix-up below)
  scan_local_kernel<<<ctg((size_t)Bn * NCH * MD), 256, 0, stream>>>(pred32, carry, ff, Bn, S, MD, CH);
  scan_carry_kernel<<<ctg((size_t)Bn * MD), 256, 0, stream>>>(carry, epref, ff, Bn, NCH, MD, CH);
  // rs = cast16(retrieved + scanned)  (fix-up fused)
  combine_fix_kernel<<<4096, 256, 0, stream>>>(ret32, pred32, epref, ff, rs16, Bn, S, MD, CH);
  // out = rs @ Wu + bu
  gemm_wmma_kernel<EPI_F32, 0><<<g_d, blk, 0, stream>>>(rs16, WuT, bu, (float*)d_out, NT, D, MD);
}
